// mortality_ViT_11656541241410
// MI455X (gfx1250) — compile-verified, matrix operand order checked by reading, not run on hardware
//
#include <hip/hip_runtime.h>

// ---------------------------------------------------------------------------
// Transformer block, B=8, N=2048, S=2049, D=512, H=8, DH=64, NC=64.
// All matmuls on V_WMMA_F32_16X16X32_BF16 (fp32 accumulate).  Every bf16
// operand buffer is padded to SP=2080 rows and 16B-aligned rows so the GEMM
// inner loop is pure global_load_b128 + v_wmma (no guards, no saveexec).
// Softmax is single-pass (row resident in registers): 2.1 GB instead of
// 6.4 GB of HBM traffic on the 1.07 GB attention tensor.
// ---------------------------------------------------------------------------
#define B_   8
#define N_   2048
#define S_   2049
#define SP_  2080   // S padded to a multiple of 32 (tile-load extent safe)
#define DIN_ 512
#define D_   512
#define H_   8
#define DH_  64
#define NC_  64
#define SCALE_ 0.125f  // 1/sqrt(64)

typedef long long i64;
typedef __attribute__((ext_vector_type(16))) __bf16 v16bf;
typedef __attribute__((ext_vector_type(8)))  __bf16 v8bf;
typedef __attribute__((ext_vector_type(8)))  float  v8f;

// ---------------------------------------------------------------------------
// Fragment loaders (wave32 WMMA 16-bit layout):
//   lane l: row m = l&15, K-half base = 8*(l>>4); element j -> k = kbase +
//   (j<8 ? j : j+8).  Each lane reads two contiguous 8-element groups at
//   [kbase, kbase+8) and [kbase+16, kbase+24).
// bf16 fast path: two 16-byte vector loads (global_load_b128).
// ---------------------------------------------------------------------------
__device__ __forceinline__ v16bf ld_frag_fast(const __bf16* __restrict__ base,
                                              i64 ld, int row, int col, int /*Mclamp*/) {
  const int lane = threadIdx.x & 31;
  const __bf16* p = base + (i64)(row + (lane & 15)) * ld + (col + ((lane >> 4) << 3));
  const v8bf lo = *reinterpret_cast<const v8bf*>(p);
  const v8bf hi = *reinterpret_cast<const v8bf*>(p + 16);
  return __builtin_shufflevector(lo, hi, 0, 1, 2, 3, 4, 5, 6, 7,
                                 8, 9, 10, 11, 12, 13, 14, 15);
}

// fp32 fast path (PV-GEMM A operand = softmaxed attention in d_out):
// unguarded scalar loads, row clamped so the last tile stays inside the panel.
__device__ __forceinline__ v16bf ld_frag_fast(const float* __restrict__ base,
                                              i64 ld, int row, int col, int Mclamp) {
  const int lane = threadIdx.x & 31;
  int m = row + (lane & 15);
  m = (m < Mclamp) ? m : (Mclamp - 1);
  const float* p = base + (i64)m * ld + (col + ((lane >> 4) << 3));
  v16bf f;
#pragma unroll
  for (int j = 0; j < 8; ++j) { f[j] = (__bf16)p[j]; f[j + 8] = (__bf16)p[j + 16]; }
  return f;
}

// Branchless ragged-K tail: clamp addresses (always valid), select value to 0.
template <typename T>
__device__ __forceinline__ v16bf ld_frag_tail(const T* __restrict__ base, i64 ld,
                                              int row, int col, int Mclamp, int K) {
  const int lane = threadIdx.x & 31;
  int m = row + (lane & 15);
  m = (m < Mclamp) ? m : (Mclamp - 1);
  const int kbase = col + ((lane >> 4) << 3);
  const T* p = base + (i64)m * ld;
  v16bf f;
#pragma unroll
  for (int j = 0; j < 16; ++j) {
    const int k = kbase + ((j < 8) ? j : (j + 8));
    const int kc = (k < K) ? k : 0;           // clamped, valid address
    const float v = (float)p[kc];
    f[j] = (__bf16)((k < K) ? v : 0.0f);      // v_cndmask, no branches
  }
  return f;
}

// C/D 16x16 fp32 layout: VGPR r, lane l -> m = r + 8*(l>>4), n = l&15.
// Epilogue can emit fp32 and/or bf16 (either pointer may be null).
__device__ __forceinline__ void store_frag(float* __restrict__ Cf, i64 ldf,
                                           __bf16* __restrict__ Cb, i64 ldbb,
                                           int row, int col, int M, int N,
                                           const v8f& c, const float* __restrict__ bias,
                                           float alpha, int relu) {
  const int lane = threadIdx.x & 31;
  const int n = col + (lane & 15);
  if (n >= N) return;
  const int m0 = row + ((lane >> 4) << 3);
  const float bb = bias ? bias[n] : 0.0f;
#pragma unroll
  for (int r = 0; r < 8; ++r) {
    const int m = m0 + r;
    if (m < M) {
      float v = c[r] * alpha + bb;
      if (relu) v = fmaxf(v, 0.0f);
      if (Cf) Cf[(i64)m * ldf + n] = v;
      if (Cb) Cb[(i64)m * ldbb + n] = (__bf16)v;
    }
  }
}

// ---------------------------------------------------------------------------
// NT-GEMM: C[z][m][n] = alpha * sum_k A[z][m][k] * Bt[z][n][k] (+bias[n]).
// 8 waves tiled 4(M)x2(N), 32x32 macro-tile per wave -> 4 v_wmma / K-step
// with a0/a1/b0/b1 fragment reuse.  Block tile 128x64.  unroll-2 pipelines
// loads under WMMA; global_prefetch pulls the K+64 lines while computing.
// ---------------------------------------------------------------------------
template <typename TA>
__global__ __launch_bounds__(256) void
gemm_nt_wmma(const TA* __restrict__ A, i64 lda, i64 sAz, int Mclamp,
             const __bf16* __restrict__ Bt, i64 ldb, i64 sBz,
             float* __restrict__ Cf, i64 ldcf, i64 sCfO, i64 sCfI,
             __bf16* __restrict__ Cb, i64 ldcb, i64 sCbO, i64 sCbI, int cDiv,
             const float* __restrict__ bias,
             int M, int N, int K, float alpha, int relu) {
  const int z = blockIdx.z;
  const int w = threadIdx.x >> 5;                 // wave id (uniform in wave)
  const int row0 = blockIdx.x * 128 + (w & 3) * 32;
  const int col0 = blockIdx.y * 64  + (w >> 2) * 32;
  if (row0 >= M || col0 >= N) return;             // wave-uniform exit

  const TA* Az = A + (i64)z * sAz;
  const __bf16* Bz = Bt + (i64)z * sBz;
  const i64 cO = (i64)(z / cDiv), cI = (i64)(z % cDiv);
  float*  Czf = Cf ? Cf + cO * sCfO + cI * sCfI : nullptr;
  __bf16* Czb = Cb ? Cb + cO * sCbO + cI * sCbI : nullptr;

  const int lane = threadIdx.x & 31;
  v8f c00 = {}, c01 = {}, c10 = {}, c11 = {};
  const int kmain = K & ~31;
#pragma unroll 2
  for (int kk = 0; kk < kmain; kk += 32) {
    v16bf a0 = ld_frag_fast(Az, lda, row0,      kk, Mclamp);
    v16bf a1 = ld_frag_fast(Az, lda, row0 + 16, kk, Mclamp);
    v16bf b0 = ld_frag_fast(Bz, ldb, col0,      kk, 1 << 30);
    v16bf b1 = ld_frag_fast(Bz, ldb, col0 + 16, kk, 1 << 30);
    if (kk + 64 <= kmain) {  // speculative prefetch of the K+64 lines
      __builtin_prefetch(Az + (i64)(row0 + lane) * lda + (kk + 64), 0, 1);
      __builtin_prefetch(Bz + (i64)(col0 + (lane & 15)) * ldb + (kk + 64), 0, 1);
    }
    c00 = __builtin_amdgcn_wmma_f32_16x16x32_bf16(false, a0, false, b0, (short)0, c00, false, false);
    c01 = __builtin_amdgcn_wmma_f32_16x16x32_bf16(false, a0, false, b1, (short)0, c01, false, false);
    c10 = __builtin_amdgcn_wmma_f32_16x16x32_bf16(false, a1, false, b0, (short)0, c10, false, false);
    c11 = __builtin_amdgcn_wmma_f32_16x16x32_bf16(false, a1, false, b1, (short)0, c11, false, false);
  }
  if (kmain < K) {                                // wave-uniform ragged-K tail
    v16bf a0 = ld_frag_tail(Az, lda, row0,      kmain, Mclamp, K);
    v16bf a1 = ld_frag_tail(Az, lda, row0 + 16, kmain, Mclamp, K);
    v16bf b0 = ld_frag_tail(Bz, ldb, col0,      kmain, 1 << 30, K);
    v16bf b1 = ld_frag_tail(Bz, ldb, col0 + 16, kmain, 1 << 30, K);
    c00 = __builtin_amdgcn_wmma_f32_16x16x32_bf16(false, a0, false, b0, (short)0, c00, false, false);
    c01 = __builtin_amdgcn_wmma_f32_16x16x32_bf16(false, a0, false, b1, (short)0, c01, false, false);
    c10 = __builtin_amdgcn_wmma_f32_16x16x32_bf16(false, a1, false, b0, (short)0, c10, false, false);
    c11 = __builtin_amdgcn_wmma_f32_16x16x32_bf16(false, a1, false, b1, (short)0, c11, false, false);
  }
  store_frag(Czf, ldcf, Czb, ldcb, row0,      col0,      M, N, c00, bias, alpha, relu);
  store_frag(Czf, ldcf, Czb, ldcb, row0,      col0 + 16, M, N, c01, bias, alpha, relu);
  store_frag(Czf, ldcf, Czb, ldcb, row0 + 16, col0,      M, N, c10, bias, alpha, relu);
  store_frag(Czf, ldcf, Czb, ldcb, row0 + 16, col0 + 16, M, N, c11, bias, alpha, relu);
}

// ---- weight prep: W (KxN fp32, row-major) -> W^T (NxK bf16) ---------------
__global__ void transpose_to_bf16(const float* __restrict__ W, __bf16* __restrict__ Wt,
                                  int K, int N) {
  const int n = blockIdx.x * 16 + (threadIdx.x & 15);
  const int k = blockIdx.y * 16 + (threadIdx.x >> 4);
  if (n < N && k < K) Wt[(i64)n * K + k] = (__bf16)W[(i64)k * N + n];
}

// ---- flat fp32 -> bf16 convert (4 elems/thread, vector loads) -------------
__global__ void cvt_bf16_flat(const float* __restrict__ s, __bf16* __restrict__ d, i64 n4) {
  const i64 i = (i64)blockIdx.x * 256 + threadIdx.x;
  if (i >= n4) return;
  const float4 v = reinterpret_cast<const float4*>(s)[i];
  __bf16* o = d + i * 4;
  o[0] = (__bf16)v.x; o[1] = (__bf16)v.y; o[2] = (__bf16)v.z; o[3] = (__bf16)v.w;
}

// ---- x[:,0,:] = survival_token (fp32 + bf16 mirrors) ----------------------
__global__ void fill_token(float* __restrict__ x, __bf16* __restrict__ xb,
                           const float* __restrict__ tok) {
  const int d = blockIdx.x * 256 + threadIdx.x;
  const int b = blockIdx.y;
  if (d < D_) {
    const float v = tok[d];
    x [(i64)b * S_  * D_ + d] = v;
    xb[(i64)b * SP_ * D_ + d] = (__bf16)v;
  }
}

// ---- qkv (b,S,3D) fp32 -> Qb,Kb (b,h,SP,DH) bf16, VT (b,h,DH,SP) bf16 -----
__global__ void reorder_qkv(const float* __restrict__ qkv, __bf16* __restrict__ Qb,
                            __bf16* __restrict__ Kb, __bf16* __restrict__ VT) {
  const int d = blockIdx.x * 256 + threadIdx.x;
  const int s = blockIdx.y, b = blockIdx.z;
  if (d >= D_) return;
  const int h = d >> 6, dd = d & 63;
  const float* p = qkv + (i64)(b * S_ + s) * (3 * D_);
  const i64 qoff = ((i64)(b * H_ + h) * SP_ + s) * DH_ + dd;
  Qb[qoff] = (__bf16)p[d];
  Kb[qoff] = (__bf16)p[D_ + d];
  VT[((i64)(b * H_ + h) * DH_ + dd) * SP_ + s] = (__bf16)p[2 * D_ + d];
}

// ---- single-pass masked softmax: row lives in 9 registers/thread ----------
// One read + one write of the 1.07 GB tensor (vs 3 passes = 6.4 GB).
__global__ __launch_bounds__(256) void softmax_rows(float* __restrict__ Aout,
                                                    const float* __restrict__ mask) {
  __shared__ float red[256];
  const int q = blockIdx.x, h = blockIdx.y, b = blockIdx.z;
  float* row = Aout + ((i64)(b * H_ + h) * S_ + q) * S_;
  const float* mb = mask + b * N_;
  const int t = threadIdx.x;
  float vals[9];
  float mx = -3.4e38f;
#pragma unroll
  for (int i = 0; i < 9; ++i) {
    const int j = t + i * 256;
    const int jc = (j < S_) ? j : (S_ - 1);     // clamped, valid address
    float v = row[jc];
    // pad[0]=false (token); token column never masked
    const float mv = (j > 0 && j < S_) ? mb[jc - 1] : 0.0f;
    v = (mv != 0.0f) ? -1.0e9f : v;
    v = (j < S_) ? v : -3.4e38f;
    vals[i] = v;
    mx = fmaxf(mx, v);
  }
  red[t] = mx; __syncthreads();
  for (int o = 128; o > 0; o >>= 1) { if (t < o) red[t] = fmaxf(red[t], red[t + o]); __syncthreads(); }
  mx = red[0]; __syncthreads();
  float sm = 0.0f;
#pragma unroll
  for (int i = 0; i < 9; ++i) {
    const int j = t + i * 256;
    const float e = (j < S_) ? __expf(vals[i] - mx) : 0.0f;
    vals[i] = e; sm += e;
  }
  red[t] = sm; __syncthreads();
  for (int o = 128; o > 0; o >>= 1) { if (t < o) red[t] += red[t + o]; __syncthreads(); }
  const float inv = 1.0f / red[0];
#pragma unroll
  for (int i = 0; i < 9; ++i) {
    const int j = t + i * 256;
    if (j < S_) row[j] = vals[i] * inv;
  }
}

// ---- Out = LayerNorm(X + Y) * g + beta; optional bf16 mirror (padded) -----
__global__ __launch_bounds__(256) void add_layernorm(const float* __restrict__ X,
                                                     const float* __restrict__ Y,
                                                     float* __restrict__ Out,
                                                     __bf16* __restrict__ OutB,
                                                     const float* __restrict__ g,
                                                     const float* __restrict__ beta) {
  __shared__ float rs[256], rs2[256];
  const int s = blockIdx.x, b = blockIdx.y, t = threadIdx.x;
  const i64 rowF = (i64)(b * S_ + s) * D_;
  const float* x = X + rowF;
  const float* y = Y + rowF;
  const float v0 = x[t] + y[t];
  const float v1 = x[t + 256] + y[t + 256];
  rs[t] = v0 + v1; rs2[t] = v0 * v0 + v1 * v1;
  __syncthreads();
  for (int off = 128; off > 0; off >>= 1) {
    if (t < off) { rs[t] += rs[t + off]; rs2[t] += rs2[t + off]; }
    __syncthreads();
  }
  const float m   = rs[0] * (1.0f / D_);
  const float var = rs2[0] * (1.0f / D_) - m * m;
  const float inv = rsqrtf(var + 1e-5f);
  const float o0 = (v0 - m) * inv * g[t]       + beta[t];
  const float o1 = (v1 - m) * inv * g[t + 256] + beta[t + 256];
  Out[rowF + t] = o0; Out[rowF + t + 256] = o1;
  if (OutB) {
    const i64 rowB = (i64)(b * SP_ + s) * D_;
    OutB[rowB + t] = (__bf16)o0; OutB[rowB + t + 256] = (__bf16)o1;
  }
}

// ---- score = x2[:,1:,:] @ W_head + b_head ; also transposed output --------
__global__ __launch_bounds__(256) void head_score(const float* __restrict__ X2,
                                                  const float* __restrict__ Wh,
                                                  const float* __restrict__ bh,
                                                  float* __restrict__ score,
                                                  float* __restrict__ outScoreT) {
  __shared__ float red[256];
  const int n = blockIdx.x, b = blockIdx.y, t = threadIdx.x;
  const float* x = X2 + ((i64)b * S_ + n + 1) * D_;
  red[t] = x[t] * Wh[t] + x[t + 256] * Wh[t + 256];
  __syncthreads();
  for (int o = 128; o > 0; o >>= 1) { if (t < o) red[t] += red[t + o]; __syncthreads(); }
  if (t == 0) {
    const float sc = red[0] + bh[0];
    score[b * N_ + n]     = sc;
    outScoreT[n * B_ + b] = sc;   // (N, B, 1)
  }
}

// ---- per-batch cluster scatter stats + survival risk ----------------------
__global__ __launch_bounds__(256) void cluster_stats(const int* __restrict__ cluster,
                                                     const float* __restrict__ score,
                                                     const float* __restrict__ risk,
                                                     float* __restrict__ outSurv,
                                                     float* __restrict__ outRisk,
                                                     float* __restrict__ outMean,
                                                     float* __restrict__ outPct) {
  __shared__ float cnt[NC_ + 1], sm[NC_ + 1];
  __shared__ float total;
  const int b = blockIdx.x, t = threadIdx.x;
  if (t < NC_ + 1) { cnt[t] = 0.0f; sm[t] = 0.0f; }
  __syncthreads();
  for (int n = t; n < N_; n += 256) {
    const int c = cluster[b * N_ + n];
    atomicAdd(&cnt[c], 1.0f);
    atomicAdd(&sm[c], score[b * N_ + n]);
  }
  __syncthreads();
  if (t == 0) { float tt = 0.0f; for (int c = 0; c < NC_; ++c) tt += cnt[c]; total = tt; }
  __syncthreads();
  if (t < NC_) {
    const float pc = cnt[t] / total;
    const float mn = (cnt[t] > 0.0f) ? sm[t] / fmaxf(cnt[t], 1.0f) : 0.0f;
    outPct[b * NC_ + t]  = pc;
    outMean[b * NC_ + t] = mn;
    if (b == 0) outRisk[t] = risk[t];
  }
  __syncthreads();
  if (t == 0) {
    float sr = 0.0f;
    for (int c = 0; c < NC_; ++c) {
      const float pc = cnt[c] / total;
      const float mn = (cnt[c] > 0.0f) ? sm[c] / fmaxf(cnt[c], 1.0f) : 0.0f;
      sr += pc * risk[c] * mn;
    }
    outSurv[b] = sr;
  }
}

// ---------------------------------------------------------------------------
extern "C" void kernel_launch(void* const* d_in, const int* in_sizes, int n_in,
                              void* d_out, int out_size, void* d_ws, size_t ws_size,
                              hipStream_t stream) {
  (void)in_sizes; (void)n_in; (void)out_size; (void)ws_size;

  const float* pe    = (const float*)d_in[0];
  const float* kmask = (const float*)d_in[2];
  const int*   clus  = (const int*)  d_in[3];
  const float* Wproj = (const float*)d_in[4];
  const float* bproj = (const float*)d_in[5];
  const float* tok   = (const float*)d_in[6];
  const float* Wqkv  = (const float*)d_in[7];
  const float* bqkv  = (const float*)d_in[8];
  const float* Wo    = (const float*)d_in[9];
  const float* bo    = (const float*)d_in[10];
  const float* ln1g  = (const float*)d_in[11];
  const float* ln1b  = (const float*)d_in[12];
  const float* W1    = (const float*)d_in[13];
  const float* b1    = (const float*)d_in[14];
  const float* W2    = (const float*)d_in[15];
  const float* b2    = (const float*)d_in[16];
  const float* ln2g  = (const float*)d_in[17];
  const float* ln2b  = (const float*)d_in[18];
  const float* Whead = (const float*)d_in[19];
  const float* bhead = (const float*)d_in[20];
  const float* risk  = (const float*)d_in[21];

  // Output layout (flat, return order).
  float* out = (float*)d_out;
  float* oSurv  = out;
  float* oScore = out + B_;
  float* oA     = oScore + (i64)N_ * B_;
  const i64 A_ELEMS = (i64)B_ * H_ * S_ * S_;
  float* oRisk  = oA + A_ELEMS;
  float* oMean  = oRisk + NC_;
  float* oPct   = oMean + (i64)B_ * NC_;

  // Workspace carve-out (256B aligned).
  char* ws = (char*)d_ws;
  size_t off = 0;
  auto carve = [&](size_t bytes) { char* p = ws + off; off = (off + bytes + 255) & ~(size_t)255; return p; };
  const i64 BSD  = (i64)B_ * S_  * D_;
  const i64 BSPD = (i64)B_ * SP_ * D_;
  __bf16* pe_bf   = (__bf16*)carve((i64)B_ * N_ * DIN_ * 2);
  float*  x       = (float*) carve(BSD * 4);
  __bf16* x_bf    = (__bf16*)carve(BSPD * 2);
  float*  qkv     = (float*) carve(BSD * 3 * 4);
  __bf16* Qb      = (__bf16*)carve((i64)B_ * H_ * SP_ * DH_ * 2);
  __bf16* Kb      = (__bf16*)carve((i64)B_ * H_ * SP_ * DH_ * 2);
  __bf16* VT      = (__bf16*)carve((i64)B_ * H_ * DH_ * SP_ * 2);
  __bf16* attnb   = (__bf16*)carve(BSPD * 2);            // P@V, bf16 only
  float*  attnpr  = (float*) carve(BSD * 4);
  float*  x1      = (float*) carve(BSD * 4);
  __bf16* x1_bf   = (__bf16*)carve(BSPD * 2);
  __bf16* ffn1    = (__bf16*)carve(BSPD * 4 * 2);        // relu hidden, bf16 only
  float*  ffn2    = (float*) carve(BSD * 4);
  float*  x2      = (float*) carve(BSD * 4);
  float*  score   = (float*) carve((i64)B_ * N_ * 4);
  __bf16* WprojT  = (__bf16*)carve((i64)D_ * DIN_ * 2);
  __bf16* WqkvT   = (__bf16*)carve((i64)3 * D_ * D_ * 2);
  __bf16* WoT     = (__bf16*)carve((i64)D_ * D_ * 2);
  __bf16* W1T     = (__bf16*)carve((i64)4 * D_ * D_ * 2);
  __bf16* W2T     = (__bf16*)carve((i64)D_ * 4 * D_ * 2);

  const dim3 blk(256);
  const int BIG = 1 << 30;

  // 1) Weight + input prep.
  transpose_to_bf16<<<dim3(D_/16,   DIN_/16), blk, 0, stream>>>(Wproj, WprojT, DIN_, D_);
  transpose_to_bf16<<<dim3(3*D_/16, D_/16),   blk, 0, stream>>>(Wqkv,  WqkvT,  D_, 3*D_);
  transpose_to_bf16<<<dim3(D_/16,   D_/16),   blk, 0, stream>>>(Wo,    WoT,    D_, D_);
  transpose_to_bf16<<<dim3(4*D_/16, D_/16),   blk, 0, stream>>>(W1,    W1T,    D_, 4*D_);
  transpose_to_bf16<<<dim3(D_/16, 4*D_/16),   blk, 0, stream>>>(W2,    W2T,    4*D_, D_);
  {
    const i64 n4 = (i64)B_ * N_ * DIN_ / 4;
    cvt_bf16_flat<<<dim3((unsigned)((n4 + 255) / 256)), blk, 0, stream>>>(pe, pe_bf, n4);
  }

  // 2) x = [token ; pe @ Wproj + bproj]   (dual fp32 + bf16 emit)
  fill_token<<<dim3(2, B_), blk, 0, stream>>>(x, x_bf, tok);
  gemm_nt_wmma<__bf16><<<dim3(16, 8, B_), blk, 0, stream>>>(
      pe_bf, DIN_, (i64)N_ * DIN_, BIG, WprojT, DIN_, 0,
      x + D_, D_, (i64)S_ * D_, 0,
      x_bf + D_, D_, (i64)SP_ * D_, 0, 1, bproj, N_, D_, DIN_, 1.0f, 0);

  // 3) qkv = x @ Wqkv + bqkv
  gemm_nt_wmma<__bf16><<<dim3(17, 24, B_), blk, 0, stream>>>(
      x_bf, D_, (i64)SP_ * D_, BIG, WqkvT, D_, 0,
      qkv, 3 * D_, (i64)S_ * 3 * D_, 0,
      (__bf16*)nullptr, 0, 0, 0, 1, bqkv, S_, 3 * D_, D_, 1.0f, 0);

  // 4) Split into per-(b,h) padded bf16 panels (V transposed for NT-GEMM).
  reorder_qkv<<<dim3(2, S_, B_), blk, 0, stream>>>(qkv, Qb, Kb, VT);

  // 5) scores = SCALE * Q Kᵀ -> straight into the A output region (fp32).
  gemm_nt_wmma<__bf16><<<dim3(17, 33, B_ * H_), blk, 0, stream>>>(
      Qb, DH_, (i64)SP_ * DH_, BIG, Kb, DH_, (i64)SP_ * DH_,
      oA, S_, (i64)S_ * S_, 0,
      (__bf16*)nullptr, 0, 0, 0, 1, nullptr, S_, S_, DH_, SCALE_, 0);

  // 6) single-pass masked softmax in place.
  softmax_rows<<<dim3(S_, H_, B_), blk, 0, stream>>>(oA, kmask);

  // 7) attn = A @ V (ragged K=2049 -> branchless tail), bf16-only output
  //    scattered into (b, SP, D) merged-head layout.
  gemm_nt_wmma<float><<<dim3(17, 1, B_ * H_), blk, 0, stream>>>(
      oA, S_, (i64)S_ * S_, S_, VT, SP_, (i64)DH_ * SP_,
      (float*)nullptr, 0, 0, 0,
      attnb, D_, (i64)SP_ * D_, DH_, H_, nullptr, S_, DH_, S_, 1.0f, 0);

  // 8) attn @ Wo + bo
  gemm_nt_wmma<__bf16><<<dim3(17, 8, B_), blk, 0, stream>>>(
      attnb, D_, (i64)SP_ * D_, BIG, WoT, D_, 0,
      attnpr, D_, (i64)S_ * D_, 0,
      (__bf16*)nullptr, 0, 0, 0, 1, bo, S_, D_, D_, 1.0f, 0);

  // 9) x1 = LN(x + attn_out)  (fp32 + bf16)
  add_layernorm<<<dim3(S_, B_), blk, 0, stream>>>(x, attnpr, x1, x1_bf, ln1g, ln1b);

  // 10) FFN: relu(x1 @ W1 + b1) @ W2 + b2
  gemm_nt_wmma<__bf16><<<dim3(17, 32, B_), blk, 0, stream>>>(
      x1_bf, D_, (i64)SP_ * D_, BIG, W1T, D_, 0,
      (float*)nullptr, 0, 0, 0,
      ffn1, 4 * D_, (i64)SP_ * 4 * D_, 0, 1, b1, S_, 4 * D_, D_, 1.0f, 1);
  gemm_nt_wmma<__bf16><<<dim3(17, 8, B_), blk, 0, stream>>>(
      ffn1, 4 * D_, (i64)SP_ * 4 * D_, BIG, W2T, 4 * D_, 0,
      ffn2, D_, (i64)S_ * D_, 0,
      (__bf16*)nullptr, 0, 0, 0, 1, b2, S_, D_, 4 * D_, 1.0f, 0);

  // 11) x2 = LN(x1 + ffn)
  add_layernorm<<<dim3(S_, B_), blk, 0, stream>>>(x1, ffn2, x2, (__bf16*)nullptr, ln2g, ln2b);

  // 12) per-token score + transposed score output
  head_score<<<dim3(N_, B_), blk, 0, stream>>>(x2, Whead, bhead, score, oScore);

  // 13) cluster statistics, survival risk, risk copy
  cluster_stats<<<dim3(B_), blk, 0, stream>>>(clus, score, risk, oSurv, oRisk, oMean, oPct);
}
